// PET_273
// MI455X (gfx1250) — matrix operand order repaired and checked by reading, not compiled, not run on hardware
//
#include <hip/hip_runtime.h>
#include <hip/hip_fp16.h>

// Problem constants (match reference)
#define U_N   200000
#define I_N   100000
#define B_N   50000
#define DIM   64
#define BTB   4096
#define E_UI  4000000
#define E_UB  2000000
#define E_BI  2000000
#define INV_TEMP 4.0f     // 1/0.25
#define ALPHA_C  0.5f

typedef __attribute__((ext_vector_type(16))) _Float16 v16h;
typedef __attribute__((ext_vector_type(8)))  _Float16 v8h;
typedef __attribute__((ext_vector_type(8)))  float    v8f;

__device__ __forceinline__ void atomAddF(float* p, float v) {
  // native global_atomic_add_f32, relaxed, device scope (no CAS loop)
  __hip_atomic_fetch_add(p, v, __ATOMIC_RELAXED, __HIP_MEMORY_SCOPE_AGENT);
}

// One-instruction wave32 butterfly step: ds_swizzle group-of-32,
// imm = {xor_mask[14:10], or_mask[9:5]=0, and_mask[4:0]=0x1f}
#define SWIZ_XOR(x, m) \
  __int_as_float(__builtin_amdgcn_ds_swizzle(__float_as_int(x), (((m) << 10) | 0x1f)))

// ---------------------------------------------------------------------------
// COO SpMM scatter-add: dst[dst_idx[e]] += vals[e] * src[src_idx[e]]
// One wave32 per edge; lane owns dims {2l, 2l+1} -> coalesced 8B/lane.
// Edge index forced wave-uniform via readfirstlane so the 3 per-edge scalars
// (val, dst row, src row) become SMEM s_load instead of 32 redundant VMEM
// lane loads. dst must be zeroed beforehand.
// ---------------------------------------------------------------------------
__global__ void __launch_bounds__(256)
spmm_edges(const float* __restrict__ vals,
           const int*   __restrict__ dst_idx,
           const int*   __restrict__ src_idx,
           const float* __restrict__ src,
           float*       __restrict__ dst, int nE) {
  const int lane = threadIdx.x & 31;
  const int wave = (blockIdx.x * blockDim.x + threadIdx.x) >> 5;
  const int nw   = (gridDim.x * blockDim.x) >> 5;
  for (int e0 = wave; e0 < nE; e0 += nw) {
    const int   e = __builtin_amdgcn_readfirstlane(e0);   // wave-uniform
    const int   r = dst_idx[e];
    const int   c = src_idx[e];
    const float v = vals[e];
    const float2 x = *(const float2*)(src + (size_t)c * DIM + 2 * lane);
    float* d = dst + (size_t)r * DIM + 2 * lane;
    atomAddF(d,     v * x.x);
    atomAddF(d + 1, v * x.y);
  }
}

// acc = (feat + layer1 + acc) / (layers+1)   [layer averaging finalize]
__global__ void finalize3(float* __restrict__ acc, const float* __restrict__ feat,
                          const float* __restrict__ l1, size_t n) {
  size_t i = (size_t)blockIdx.x * blockDim.x + threadIdx.x;
  const size_t stride = (size_t)gridDim.x * blockDim.x;
  for (; i < n; i += stride)
    acc[i] = (feat[i] + l1[i] + acc[i]) * (1.0f / 3.0f);
}

// ---------------------------------------------------------------------------
// Gather rows by index, L2-normalize (rsqrt(sumsq+1e-8)), emit f16 row-major.
// One wave per row; lane owns 2 dims; sumsq via ds_swizzle butterfly.
// ---------------------------------------------------------------------------
__global__ void __launch_bounds__(256)
gather_norm_f16(const float* __restrict__ A, const float* __restrict__ B,
                const int* __restrict__ idx, int idxStride,
                _Float16* __restrict__ Ah, _Float16* __restrict__ Bh) {
  const int lane = threadIdx.x & 31;
  const int row  = (blockIdx.x * blockDim.x + threadIdx.x) >> 5;
  if (row >= BTB) return;
  const int g = idx[row * idxStride];
  float2 xa = *(const float2*)(A + (size_t)g * DIM + 2 * lane);
  float2 xb = *(const float2*)(B + (size_t)g * DIM + 2 * lane);
  float sa = xa.x * xa.x + xa.y * xa.y;
  float sb = xb.x * xb.x + xb.y * xb.y;
  sa += SWIZ_XOR(sa, 1);  sb += SWIZ_XOR(sb, 1);
  sa += SWIZ_XOR(sa, 2);  sb += SWIZ_XOR(sb, 2);
  sa += SWIZ_XOR(sa, 4);  sb += SWIZ_XOR(sb, 4);
  sa += SWIZ_XOR(sa, 8);  sb += SWIZ_XOR(sb, 8);
  sa += SWIZ_XOR(sa, 16); sb += SWIZ_XOR(sb, 16);
  const float ra = rsqrtf(sa + 1e-8f);
  const float rb = rsqrtf(sb + 1e-8f);
  Ah[(size_t)row * DIM + 2 * lane]     = (_Float16)(xa.x * ra);
  Ah[(size_t)row * DIM + 2 * lane + 1] = (_Float16)(xa.y * ra);
  Bh[(size_t)row * DIM + 2 * lane]     = (_Float16)(xb.x * rb);
  Bh[(size_t)row * DIM + 2 * lane + 1] = (_Float16)(xb.y * rb);
}

// ---------------------------------------------------------------------------
// Fused logits GEMM + exp-row-sum + diagonal, via V_WMMA_F32_16X16X32_F16.
// logits = (Ah @ Bh^T) * INV_TEMP over [4096,4096], K=64 (2 WMMA per tile).
// Logits are bounded by +-4, so raw exp (no max-shift) is numerically safe.
// A frag (16x32 f16, ISA layout): lane m=lo, chunks at K=8*hi and 16+8*hi.
// B frag (32x16 f16): lane n=lo, chunks at K=8*hi and 16+8*hi.
// C/D: VGPR r -> M=r+8*hi, N=lo.
// Row-sum over the 16 N-lanes with 4 ds_swizzle xor steps (stay within half).
// ---------------------------------------------------------------------------
__global__ void __launch_bounds__(256)
nce_gemm_lse(const _Float16* __restrict__ Ah, const _Float16* __restrict__ Bh,
             float* __restrict__ rowExpSum, float* __restrict__ diagAcc) {
  const int lane = threadIdx.x & 31;
  const int lo = lane & 15, hi = lane >> 4;
  const int TPD = BTB / 16;  // 256 tiles per dimension
  const int tile = blockIdx.x * (blockDim.x >> 5) + (threadIdx.x >> 5);
  const int tm = tile / TPD, tn = tile % TPD;

  const _Float16* arow = Ah + (size_t)(tm * 16 + lo) * DIM;
  const _Float16* brow = Bh + (size_t)(tn * 16 + lo) * DIM;

  v8f c = {};
#pragma unroll
  for (int kk = 0; kk < DIM; kk += 32) {
    const v8h a0 = *(const v8h*)(arow + kk + 8 * hi);
    const v8h a1 = *(const v8h*)(arow + kk + 16 + 8 * hi);
    const v8h b0 = *(const v8h*)(brow + kk + 8 * hi);
    const v8h b1 = *(const v8h*)(brow + kk + 16 + 8 * hi);
    v16h a, b;
#pragma unroll
    for (int t = 0; t < 8; ++t) { a[t] = a0[t]; a[t + 8] = a1[t]; b[t] = b0[t]; b[t + 8] = b1[t]; }
    c = __builtin_amdgcn_wmma_f32_16x16x32_f16(false, a, false, b,
                                               (short)0, c, false, false);
  }

#pragma unroll
  for (int r = 0; r < 8; ++r) {
    const float logit = c[r] * INV_TEMP;
    float e = __expf(logit);
    e += SWIZ_XOR(e, 1);
    e += SWIZ_XOR(e, 2);
    e += SWIZ_XOR(e, 4);
    e += SWIZ_XOR(e, 8);
    if (lo == 0) atomAddF(&rowExpSum[tm * 16 + 8 * hi + r], e);
    if (tm == tn && lo == (8 * hi + r)) atomAddF(diagAcc, logit);
  }
}

// ---------------------------------------------------------------------------
// Per-sample: tri-view dot products -> 3->3->3->1 tanh MLP -> BPR softplus.
// ---------------------------------------------------------------------------
__global__ void __launch_bounds__(256)
bpr_fusion(const float* __restrict__ u1, const float* __restrict__ u2,
           const float* __restrict__ u3, const float* __restrict__ b1,
           const float* __restrict__ b2, const float* __restrict__ b3,
           const int* __restrict__ users_idx, const int* __restrict__ bundles_idx,
           const float* __restrict__ L1w, const float* __restrict__ L1b,
           const float* __restrict__ L2w, const float* __restrict__ L2b,
           const float* __restrict__ L3w, const float* __restrict__ L3b,
           float* __restrict__ bprAcc) {
  const int t = blockIdx.x * blockDim.x + threadIdx.x;
  if (t >= BTB) return;
  const int u = users_idx[t];
  const int bp0 = bundles_idx[2 * t], bp1 = bundles_idx[2 * t + 1];
  const float* ur0 = u1 + (size_t)u * DIM;
  const float* ur1 = u2 + (size_t)u * DIM;
  const float* ur2 = u3 + (size_t)u * DIM;

  float s[2][3] = {{0.f, 0.f, 0.f}, {0.f, 0.f, 0.f}};
  for (int d = 0; d < DIM; ++d) {
    const float e0 = ur0[d], e1 = ur1[d], e2 = ur2[d];
    const size_t o0 = (size_t)bp0 * DIM + d;
    const size_t o1 = (size_t)bp1 * DIM + d;
    s[0][0] += e0 * b1[o0]; s[0][1] += e1 * b2[o0]; s[0][2] += e2 * b3[o0];
    s[1][0] += e0 * b1[o1]; s[1][1] += e1 * b2[o1]; s[1][2] += e2 * b3[o1];
  }

  float pred[2];
#pragma unroll
  for (int p = 0; p < 2; ++p) {
    float h1[3], h2[3];
#pragma unroll
    for (int i = 0; i < 3; ++i) {
      float acc = L1b[i];
      for (int j = 0; j < 3; ++j) acc += s[p][j] * L1w[i * 3 + j];
      h1[i] = tanhf(acc);
    }
#pragma unroll
    for (int i = 0; i < 3; ++i) {
      float acc = L2b[i];
      for (int j = 0; j < 3; ++j) acc += h1[j] * L2w[i * 3 + j];
      h2[i] = tanhf(acc);
    }
    float acc = L3b[0];
    for (int j = 0; j < 3; ++j) acc += h2[j] * L3w[j];
    pred[p] = acc;
  }
  const float x = -(pred[0] - pred[1]);
  const float sp = fmaxf(x, 0.0f) + log1pf(__expf(-fabsf(x)));  // softplus
  atomAddF(bprAcc, sp);
}

// scal = [diag1, diag2, bprSum]; out = bpr/BT + alpha*(nce1+nce2)
__global__ void final_combine(const float* __restrict__ rs1,
                              const float* __restrict__ rs2,
                              const float* __restrict__ scal,
                              float* __restrict__ out) {
  __shared__ float red[256];
  float local = 0.f;
  for (int i = threadIdx.x; i < BTB; i += blockDim.x)
    local += __logf(rs1[i]) + __logf(rs2[i]);
  red[threadIdx.x] = local;
  __syncthreads();
  for (int s = 128; s > 0; s >>= 1) {
    if ((int)threadIdx.x < s) red[threadIdx.x] += red[threadIdx.x + s];
    __syncthreads();
  }
  if (threadIdx.x == 0) {
    const float nce = (red[0] - scal[0] - scal[1]) / (float)BTB;
    out[0] = scal[2] / (float)BTB + ALPHA_C * nce;
  }
}

// ---------------------------------------------------------------------------
extern "C" void kernel_launch(void* const* d_in, const int* in_sizes, int n_in,
                              void* d_out, int out_size, void* d_ws, size_t ws_size,
                              hipStream_t stream) {
  (void)in_sizes; (void)n_in; (void)out_size; (void)ws_size;
  const float* users_feat   = (const float*)d_in[0];
  const float* bundles_feat = (const float*)d_in[1];
  const float* items_feat   = (const float*)d_in[2];
  const float* L1w = (const float*)d_in[3];
  const float* L1b = (const float*)d_in[4];
  const float* L2w = (const float*)d_in[5];
  const float* L2b = (const float*)d_in[6];
  const float* L3w = (const float*)d_in[7];
  const float* L3b = (const float*)d_in[8];
  const float* ui_val = (const float*)d_in[9];
  const float* ub_val = (const float*)d_in[10];
  const float* bi_val = (const float*)d_in[11];
  const int* ui_row = (const int*)d_in[12];
  const int* ui_col = (const int*)d_in[13];
  const int* ub_row = (const int*)d_in[14];
  const int* ub_col = (const int*)d_in[15];
  const int* bi_row = (const int*)d_in[16];
  const int* bi_col = (const int*)d_in[17];
  const int* users_idx   = (const int*)d_in[18];
  const int* bundles_idx = (const int*)d_in[19];

  // ---- workspace carve (256B aligned regions) ----
  char* base = (char*)d_ws;
  size_t off = 0;
  auto carve = [&](size_t bytes) -> char* {
    char* p = base + off;
    off += (bytes + 255) & ~(size_t)255;
    return p;
  };
  const size_t Ub = (size_t)U_N * DIM * 4, Ib = (size_t)I_N * DIM * 4,
               Bb = (size_t)B_N * DIM * 4;
  float* u1 = (float*)carve(Ub);
  float* i1 = (float*)carve(Ib);
  float* u2 = (float*)carve(Ub);
  float* b2 = (float*)carve(Bb);
  float* b1 = (float*)carve(Bb);
  float* tU = (float*)carve(Ub);   // layer-1 temp (U side); later reused as u3
  float* tI = (float*)carve(Ib);   // layer-1 temp (I/B side); later reused as b3
  _Float16* a1h = (_Float16*)carve((size_t)BTB * DIM * 2);
  _Float16* b1h = (_Float16*)carve((size_t)BTB * DIM * 2);
  _Float16* a2h = (_Float16*)carve((size_t)BTB * DIM * 2);
  _Float16* b2h = (_Float16*)carve((size_t)BTB * DIM * 2);
  float* rs1  = (float*)carve(BTB * 4);
  float* rs2  = (float*)carve(BTB * 4);
  float* scal = (float*)carve(4 * 4);   // [diag1, diag2, bprSum]

  const int SPB = 2048, SPT = 256;      // spmm grid: 16384 waves

  // ---- View 1: u-i Laplacian propagation (2 layers) ----
  hipMemsetAsync(tU, 0, Ub, stream);
  hipMemsetAsync(tI, 0, Ib, stream);
  hipMemsetAsync(u1, 0, Ub, stream);
  hipMemsetAsync(i1, 0, Ib, stream);
  spmm_edges<<<SPB, SPT, 0, stream>>>(ui_val, ui_row, ui_col, items_feat, tU, E_UI);
  spmm_edges<<<SPB, SPT, 0, stream>>>(ui_val, ui_col, ui_row, users_feat, tI, E_UI);
  spmm_edges<<<SPB, SPT, 0, stream>>>(ui_val, ui_row, ui_col, tI, u1, E_UI);
  spmm_edges<<<SPB, SPT, 0, stream>>>(ui_val, ui_col, ui_row, tU, i1, E_UI);
  finalize3<<<4096, 256, 0, stream>>>(u1, users_feat, tU, (size_t)U_N * DIM);
  finalize3<<<4096, 256, 0, stream>>>(i1, items_feat, tI, (size_t)I_N * DIM);

  // b1 = segsum(bi, i1)
  hipMemsetAsync(b1, 0, Bb, stream);
  spmm_edges<<<SPB, SPT, 0, stream>>>(bi_val, bi_row, bi_col, i1, b1, E_BI);

  // ---- View 2: u-b Laplacian propagation (2 layers) ----
  hipMemsetAsync(tU, 0, Ub, stream);
  hipMemsetAsync(tI, 0, Bb, stream);
  hipMemsetAsync(u2, 0, Ub, stream);
  hipMemsetAsync(b2, 0, Bb, stream);
  spmm_edges<<<SPB, SPT, 0, stream>>>(ub_val, ub_row, ub_col, bundles_feat, tU, E_UB);
  spmm_edges<<<SPB, SPT, 0, stream>>>(ub_val, ub_col, ub_row, users_feat, tI, E_UB);
  spmm_edges<<<SPB, SPT, 0, stream>>>(ub_val, ub_row, ub_col, tI, u2, E_UB);
  spmm_edges<<<SPB, SPT, 0, stream>>>(ub_val, ub_col, ub_row, tU, b2, E_UB);
  finalize3<<<4096, 256, 0, stream>>>(u2, users_feat, tU, (size_t)U_N * DIM);
  finalize3<<<4096, 256, 0, stream>>>(b2, bundles_feat, tI, (size_t)B_N * DIM);

  // ---- View 3: one-hop raw aggregation (reuse temps) ----
  float* u3 = tU; float* b3 = tI;
  hipMemsetAsync(u3, 0, Ub, stream);
  hipMemsetAsync(b3, 0, Bb, stream);
  spmm_edges<<<SPB, SPT, 0, stream>>>(ui_val, ui_row, ui_col, items_feat, u3, E_UI);
  spmm_edges<<<SPB, SPT, 0, stream>>>(bi_val, bi_row, bi_col, items_feat, b3, E_BI);

  // ---- InfoNCE (WMMA logits GEMM fused with exp-row-sum) ----
  hipMemsetAsync(rs1, 0, BTB * 4, stream);
  hipMemsetAsync(rs2, 0, BTB * 4, stream);
  hipMemsetAsync(scal, 0, 4 * 4, stream);
  gather_norm_f16<<<(BTB * 32) / 256, 256, 0, stream>>>(u1, u2, users_idx, 1, a1h, b1h);
  gather_norm_f16<<<(BTB * 32) / 256, 256, 0, stream>>>(b1, b2, bundles_idx, 2, a2h, b2h);
  const int TILES = (BTB / 16) * (BTB / 16);  // 65536 tiles, 8 waves/block
  nce_gemm_lse<<<TILES / 8, 256, 0, stream>>>(a1h, b1h, rs1, &scal[0]);
  nce_gemm_lse<<<TILES / 8, 256, 0, stream>>>(a2h, b2h, rs2, &scal[1]);

  // ---- BPR + final combine ----
  bpr_fusion<<<BTB / 256, 256, 0, stream>>>(u1, u2, u3, b1, b2, b3,
                                            users_idx, bundles_idx,
                                            L1w, L1b, L2w, L2b, L3w, L3b,
                                            &scal[2]);
  final_combine<<<1, 256, 0, stream>>>(rs1, rs2, scal, (float*)d_out);
}
